// Interaction_Block_68504728371701
// MI455X (gfx1250) — matrix operand order repaired and checked by reading, not compiled, not run on hardware
//
#include <hip/hip_runtime.h>
#include <hip/hip_bf16.h>
#include <math.h>
#include <stdint.h>

typedef __bf16 bf16_t;
typedef __attribute__((ext_vector_type(16))) __bf16 v16bf;
typedef __attribute__((ext_vector_type(8)))  __bf16 v8bf;
typedef __attribute__((ext_vector_type(8)))  float  v8f;

union Frag { v16bf v; v8bf h[2]; };

#define WMMA_BF16(a,b,c) __builtin_amdgcn_wmma_f32_16x16x32_bf16(false,(a),false,(b),(short)0,(c),false,false)

struct P {
  const int*   edge_index;     // [2,E]
  const float *senders_pos, *receivers_pos, *edge_dx, *edge_attr;
  const float *va, *vb, *vc, *svt, *swt, *rvt, *rwt;
  const float *node_latent;    // [N,128]
  const int*   node_type;      // [N,2]
  const float *node_weights;   // [N,1]
  const float *ef_w1,*ef_b1,*ef_b2,*ef_g,*ef_be;
  const float *nf_w1,*nf_b1,*nf_b2,*nf_g,*nf_be;
  const float *it_b1,*it_b2,*it_g,*it_be;
  const float *i1_b1,*i1_w2,*i1_b2;
  const float *i2_b1,*i2_w2,*i2_b2;
  const float *fs_b1,*fs_w2,*fs_b2;
  const float *ex_b1,*ex_w2,*ex_b2;
  const float *im_b1,*im_w2,*im_b2;
  const float *ii_b1,*ii_w2,*ii_b2;
  const bf16_t *WTnf2,*WTef2,*WTint1,*WTint2,*WTi1,*WTi2,*WTfsc,*WText,*WTim,*WTii;
  float *cnt,*s_f,*s_a,*M_g,*dv_sum,*netF,*netT,*dv_raw,*dv_ext,*im_dt,*ii_dt;
  float *fij,*aij,*lam,*wbg;
  float *out_dv,*out_dw,*out_inter;
  int N, E;
};

__device__ __forceinline__ v8f splat8(float x){ v8f v={x,x,x,x,x,x,x,x}; return v; }

// ---- fragment loaders (ISA 7.12.2 layouts) ----
__device__ __forceinline__ v16bf ldA_lds(const bf16_t* row, int ks, int half){
  const int b1 = ks*32 + (half ? 8 : 0);
  Frag f;
  f.h[0] = *(const v8bf*)(row + b1);
  f.h[1] = *(const v8bf*)(row + b1 + 16);
  return f.v;
}
__device__ __forceinline__ v16bf ldA_g32(const float* row, int ks, int half){
  const int b1 = ks*32 + (half ? 8 : 0);
  Frag f;
  #pragma unroll
  for (int t=0;t<2;++t)
    #pragma unroll
    for (int j=0;j<8;++j)
      f.h[t][j] = (bf16_t)row[b1 + t*16 + j];
  return f.v;
}
__device__ __forceinline__ v16bf ldA_g32_sum(const float* r0, const float* r1, int ks, int half){
  const int b1 = ks*32 + (half ? 8 : 0);
  Frag f;
  #pragma unroll
  for (int t=0;t<2;++t)
    #pragma unroll
    for (int j=0;j<8;++j)
      f.h[t][j] = (bf16_t)(r0[b1+t*16+j] + r1[b1+t*16+j]);
  return f.v;
}
__device__ __forceinline__ v16bf ldB(const bf16_t* col, int ks, int half){
  const int b1 = ks*32 + (half ? 8 : 0);
  Frag f;
  f.h[0] = *(const v8bf*)(col + b1);
  f.h[1] = *(const v8bf*)(col + b1 + 16);
  return f.v;
}

// ---- CDNA5 async weight staging: global -> LDS, ASYNCcnt-tracked ----
__device__ __forceinline__ void async_cp16(bf16_t* lds_dst, const bf16_t* gsrc){
  asm volatile("global_load_async_to_lds_b128 %0, %1, off"
               :: "v"((uint32_t)(size_t)lds_dst), "v"(gsrc) : "memory");
}
__device__ __forceinline__ void wait_async(){
  asm volatile("s_wait_asynccnt 0x0" ::: "memory");
}
// stage a [128][128] bf16 weight block (32KB) with 128 threads, 16 x b128 each
__device__ __forceinline__ void stage128_issue(const bf16_t* W, bf16_t* Bbuf, int tid){
  #pragma unroll
  for (int i=0;i<16;++i){
    const int c = tid + i*128;           // 2048 chunks of 8 bf16
    async_cp16(Bbuf + c*8, W + c*8);
  }
}
// stage one K=128 phase of the [128][384] inter_enc.l1 block into [col][128] layout
__device__ __forceinline__ void stage_int1_issue(const bf16_t* W, int phase, bf16_t* Bbuf, int tid){
  #pragma unroll
  for (int i=0;i<16;++i){
    const int c   = tid + i*128;
    const int col = c >> 4;              // 16 chunks per column
    const int o   = (c & 15)*8;
    async_cp16(Bbuf + col*128 + o, W + (size_t)col*384 + phase*128 + o);
  }
}

// ---- GEMM building blocks (B pointer may be LDS or global) ----
__device__ __forceinline__ void gemm_dump_f32(const bf16_t* A, const bf16_t* B,
    const float* bias, float* F, int lane, bool relu)
{
  const int half = lane>>4, n = lane&15, m = lane&15;
  #pragma unroll
  for (int sub=0; sub<2; ++sub){
    const bf16_t* arow = A + (sub*16+m)*128;
    v16bf af[4];
    #pragma unroll
    for (int ks=0; ks<4; ++ks) af[ks] = ldA_lds(arow, ks, half);
    #pragma unroll
    for (int cb=0; cb<8; ++cb){
      const bf16_t* bcol = B + (size_t)(cb*16+n)*128;
      v8f acc = splat8(bias[cb*16+n]);
      #pragma unroll
      for (int ks=0; ks<4; ++ks)
        acc = WMMA_BF16(af[ks], ldB(bcol, ks, half), acc);
      #pragma unroll
      for (int i=0;i<8;++i){
        float x = acc[i]; if (relu) x = fmaxf(x, 0.f);
        F[(sub*16 + half*8 + i)*128 + cb*16 + n] = x;
      }
    }
  }
}
// one K=128 phase accumulating into acc[2][8], A tile in LDS
__device__ __forceinline__ void phase_lds(v8f (&acc)[2][8], const bf16_t* Atile,
    const bf16_t* B, int lane)
{
  const int half = lane>>4, n = lane&15, m = lane&15;
  #pragma unroll
  for (int sub=0; sub<2; ++sub){
    v16bf af[4];
    #pragma unroll
    for (int ks=0; ks<4; ++ks) af[ks] = ldA_lds(Atile + (sub*16+m)*128, ks, half);
    #pragma unroll
    for (int cb=0; cb<8; ++cb){
      const bf16_t* bcol = B + (size_t)(cb*16+n)*128;
      #pragma unroll
      for (int ks=0; ks<4; ++ks)
        acc[sub][cb] = WMMA_BF16(af[ks], ldB(bcol, ks, half), acc[sub][cb]);
    }
  }
}
// one K=128 phase with pre-gathered register fragments
__device__ __forceinline__ void phase_regs(v8f (&acc)[2][8], const v16bf (&af)[2][4],
    const bf16_t* B, int lane)
{
  const int half = lane>>4, n = lane&15;
  #pragma unroll
  for (int sub=0; sub<2; ++sub)
    #pragma unroll
    for (int cb=0; cb<8; ++cb){
      const bf16_t* bcol = B + (size_t)(cb*16+n)*128;
      #pragma unroll
      for (int ks=0; ks<4; ++ks)
        acc[sub][cb] = WMMA_BF16(af[sub][ks], ldB(bcol, ks, half), acc[sub][cb]);
    }
}

__device__ __forceinline__ void meanvar(const float* x, float& mu, float& rstd){
  float s = 0.f;
  for (int k=0;k<128;++k) s += x[k];
  mu = s * (1.f/128.f);
  float v = 0.f;
  for (int k=0;k<128;++k){ float d = x[k]-mu; v += d*d; }
  rstd = rsqrtf(v*(1.f/128.f) + 1e-5f);
}

__device__ __forceinline__ void small_layer(const float* W1, const float* b1,
    const float* x, int fin, bf16_t* Arow)
{
  for (int j=0;j<128;++j){
    float acc = b1[j];
    for (int i=0;i<fin;++i) acc += x[i]*W1[i*128+j];
    Arow[j] = (bf16_t)fmaxf(acc, 0.f);
  }
}

__device__ __forceinline__ void node_hidden(const v16bf (&af)[2][4], const bf16_t* B,
    const float* b1, float* F, int lane)
{
  const int half = lane>>4, n = lane&15;
  #pragma unroll
  for (int sub=0; sub<2; ++sub){
    #pragma unroll
    for (int cb=0; cb<8; ++cb){
      const bf16_t* bcol = B + (size_t)(cb*16+n)*128;
      v8f acc = splat8(b1[cb*16+n]);
      #pragma unroll
      for (int ks=0; ks<4; ++ks)
        acc = WMMA_BF16(af[sub][ks], ldB(bcol, ks, half), acc);
      #pragma unroll
      for (int i=0;i<8;++i)
        F[(sub*16 + half*8 + i)*128 + cb*16 + n] = fmaxf(acc[i], 0.f);
    }
  }
}

// ---------------- node-side MLPs (ext_dv, inv_mass, inv_inertia) ----------------
// 4 waves/WG, 32 nodes per wave. Per-head weights async-staged into shared LDS,
// staging overlapped with the previous head's 128->{3,1,1} VALU dots.
// LDS: 32KB Bbuf + 4 x 16KB F = 96KB.
__global__ void __launch_bounds__(128) node_kernel(P p)
{
  extern __shared__ char smem[];
  bf16_t* Bbuf = (bf16_t*)smem;                        // 32KB shared weight stage
  const int tid  = threadIdx.x;
  const int wid  = tid >> 5;
  const int lane = tid & 31;
  float* F = (float*)(smem + 32768 + wid*16384);       // [32][128] fp32 per wave
  const int half = lane>>4, m = lane&15;
  const int t0 = (blockIdx.x*4 + wid)*32;
  const int node = t0 + lane;

  stage128_issue(p.WText, Bbuf, tid);                  // overlap: A-fragment gather

  v16bf af[2][4];
  #pragma unroll
  for (int sub=0; sub<2; ++sub){
    int nd = t0 + sub*16 + m; if (nd >= p.N) nd = p.N-1;
    const float* row = p.node_latent + (size_t)nd*128;
    #pragma unroll
    for (int ks=0; ks<4; ++ks) af[sub][ks] = ldA_g32(row, ks, half);
  }

  wait_async(); __syncthreads();
  node_hidden(af, Bbuf, p.ex_b1, F, lane);             // ext_dv hidden
  __syncthreads();
  stage128_issue(p.WTim, Bbuf, tid);                   // overlap: ext_dv dots
  if (node < p.N){
    float o[3] = {p.ex_b2[0], p.ex_b2[1], p.ex_b2[2]};
    const float* h = F + lane*128;
    for (int k=0;k<128;++k){ float hv=h[k];
      for (int c=0;c<3;++c) o[c] += hv*p.ex_w2[k*3+c]; }
    for (int c=0;c<3;++c){ p.dv_raw[node*3+c]=o[c]; p.dv_ext[node*3+c]=o[c]; }
  }
  wait_async(); __syncthreads();
  node_hidden(af, Bbuf, p.im_b1, F, lane);             // inv_mass hidden
  __syncthreads();
  stage128_issue(p.WTii, Bbuf, tid);                   // overlap: inv_mass dot
  if (node < p.N){
    float o = p.im_b2[0];
    const float* h = F + lane*128;
    for (int k=0;k<128;++k) o += h[k]*p.im_w2[k];
    p.im_dt[node] = o;
  }
  wait_async(); __syncthreads();
  node_hidden(af, Bbuf, p.ii_b1, F, lane);             // inv_inertia hidden
  __syncthreads();
  if (node < p.N){
    float o = p.ii_b2[0];
    const float* h = F + lane*128;
    for (int k=0;k<128;++k) o += h[k]*p.ii_w2[k];
    p.ii_dt[node] = o;
  }
}

// ---------------- edge pipeline pass 1 ----------------
// 4 waves / WG, 32 edges per wave. Weights staged layer-by-layer into a shared
// 32KB LDS buffer via global_load_async_to_lds_b128, overlapped with VALU work.
// LDS: 32KB Bbuf + 4 x (16KB F fp32 + 8KB U bf16 + 8KB A bf16) = 160KB.
__global__ void __launch_bounds__(128) edge_kernel(P p)
{
  extern __shared__ char smem[];
  bf16_t* Bbuf = (bf16_t*)smem;                       // 32KB shared weight stage
  const int tid  = threadIdx.x;
  const int wid  = tid >> 5;
  const int lane = tid & 31;
  char* wbase = smem + 32768 + wid*32768;
  float*  F = (float*)wbase;                          // [32][128] fp32 16KB
  bf16_t* U = (bf16_t*)(wbase + 16384);               // [32][128] bf16 8KB
  bf16_t* A = (bf16_t*)(wbase + 24576);               // [32][128] bf16 8KB

  const int e0 = (blockIdx.x*4 + wid)*32;
  const int el = e0 + lane;
  const bool valid = el < p.E;
  const int eli = valid ? el : p.E-1;

  stage128_issue(p.WTnf2, Bbuf, tid);   // overlap: per-lane features

  float va3[3],vb3[3],vc3[3],sv[3],sw[3],rv[3],rw[3],dx[3];
  #pragma unroll
  for (int d=0; d<3; ++d){
    va3[d]=p.va[eli*3+d]; vb3[d]=p.vb[eli*3+d]; vc3[d]=p.vc[eli*3+d];
    sv[d]=p.svt[eli*3+d]; sw[d]=p.swt[eli*3+d];
    rv[d]=p.rvt[eli*3+d]; rw[d]=p.rwt[eli*3+d];
    dx[d]=p.edge_dx[eli*3+d];
  }
  auto dot3 = [](const float* a, const float* b){ return a[0]*b[0]+a[1]*b[1]+a[2]*b[2]; };
  float sf6[6] = { dot3(va3,sv), dot3(vb3,sv), dot3(vc3,sv),
                   dot3(va3,sw), dot3(vb3,sw), dot3(vc3,sw) };
  float rf6[6] = {-dot3(va3,rv),-dot3(vb3,rv),-dot3(vc3,rv),
                  -dot3(va3,rw),-dot3(vb3,rw),-dot3(vc3,rw) };
  float ef4[4] = { sqrtf(dot3(dx,dx)),
                   p.edge_attr[eli*3+0], p.edge_attr[eli*3+1], p.edge_attr[eli*3+2] };

  // s_lat
  small_layer(p.nf_w1, p.nf_b1, sf6, 6, A + lane*128);
  wait_async(); __syncthreads();
  gemm_dump_f32(A, Bbuf, p.nf_b2, F, lane, false);
  __syncthreads();
  { float mu,rs_; meanvar(F+lane*128, mu, rs_);
    for (int k=0;k<128;++k)
      U[lane*128+k] = (bf16_t)((F[lane*128+k]-mu)*rs_*p.nf_g[k] + p.nf_be[k]); }
  small_layer(p.nf_w1, p.nf_b1, rf6, 6, A + lane*128);
  __syncthreads();
  // r_lat (reuses staged nf2)
  gemm_dump_f32(A, Bbuf, p.nf_b2, F, lane, false);
  __syncthreads();
  stage128_issue(p.WTef2, Bbuf, tid);     // overlap: LN add + e small layer
  { float mu,rs_; meanvar(F+lane*128, mu, rs_);
    for (int k=0;k<128;++k)
      U[lane*128+k] = (bf16_t)((float)U[lane*128+k]
                        + (F[lane*128+k]-mu)*rs_*p.nf_g[k] + p.nf_be[k]); }
  small_layer(p.ef_w1, p.ef_b1, ef4, 4, A + lane*128);
  wait_async(); __syncthreads();
  // e_lat
  gemm_dump_f32(A, Bbuf, p.ef_b2, F, lane, false);
  __syncthreads();
  stage_int1_issue(p.WTint1, 0, Bbuf, tid);   // overlap: LN(e) -> A
  { float mu,rs_; meanvar(F+lane*128, mu, rs_);
    for (int k=0;k<128;++k)
      A[lane*128+k] = (bf16_t)((F[lane*128+k]-mu)*rs_*p.ef_g[k] + p.ef_be[k]); }

  // inter_enc l1: K=384 in three staged phases, acc carried in VGPRs
  const int n = lane & 15;
  v8f acc[2][8];
  #pragma unroll
  for (int sub=0; sub<2; ++sub)
    #pragma unroll
    for (int cb=0; cb<8; ++cb) acc[sub][cb] = splat8(p.it_b1[cb*16+n]);

  wait_async(); __syncthreads();
  phase_lds(acc, U, Bbuf, lane);                       // u = LN(s)+LN(r)
  __syncthreads();
  stage_int1_issue(p.WTint1, 1, Bbuf, tid);            // overlap: gather nl_s+nl_r
  v16bf nlf[2][4];
  {
    const int half = lane>>4, m = lane&15;
    #pragma unroll
    for (int sub=0; sub<2; ++sub){
      int erow = e0 + sub*16 + m; if (erow >= p.E) erow = p.E-1;
      const float* rs = p.node_latent + (size_t)p.edge_index[erow]*128;
      const float* rr = p.node_latent + (size_t)p.edge_index[p.E+erow]*128;
      #pragma unroll
      for (int ks=0; ks<4; ++ks) nlf[sub][ks] = ldA_g32_sum(rs, rr, ks, half);
    }
  }
  wait_async(); __syncthreads();
  phase_regs(acc, nlf, Bbuf, lane);                    // node_latent[s]+node_latent[r]
  __syncthreads();
  stage_int1_issue(p.WTint1, 2, Bbuf, tid);
  wait_async(); __syncthreads();
  phase_lds(acc, A, Bbuf, lane);                       // e_lat
  __syncthreads();
  stage128_issue(p.WTint2, Bbuf, tid);                 // overlap: dump hidden
  {
    const int half = lane>>4;
    #pragma unroll
    for (int sub=0; sub<2; ++sub)
      #pragma unroll
      for (int cb=0; cb<8; ++cb)
        #pragma unroll
        for (int i=0;i<8;++i)
          A[(sub*16 + half*8 + i)*128 + cb*16 + n] = (bf16_t)fmaxf(acc[sub][cb][i], 0.f);
  }
  wait_async(); __syncthreads();
  gemm_dump_f32(A, Bbuf, p.it_b2, F, lane, false);
  __syncthreads();
  stage128_issue(p.WTi1, Bbuf, tid);                   // overlap: LN(inter)
  { float mu,rs_; meanvar(F+lane*128, mu, rs_);
    float* outp = p.out_inter + (size_t)eli*128;
    for (int k=0;k<128;++k){
      float y = (F[lane*128+k]-mu)*rs_*p.it_g[k] + p.it_be[k];
      if (valid) outp[k] = y;
      A[lane*128+k] = (bf16_t)y;
    } }
  wait_async(); __syncthreads();

  // decoder heads
  float cf[3], ca[3], lamv;
  gemm_dump_f32(A, Bbuf, p.i1_b1, F, lane, true);
  __syncthreads();
  stage128_issue(p.WTi2, Bbuf, tid);                   // overlap: i1 head dots
  {
    const float* h = F + lane*128;
    for (int c=0;c<3;++c) cf[c] = p.i1_b2[c];
    for (int k=0;k<128;++k){ float hv=h[k];
      for (int c=0;c<3;++c) cf[c] += hv*p.i1_w2[k*3+c]; }
  }
  wait_async(); __syncthreads();
  gemm_dump_f32(A, Bbuf, p.i2_b1, F, lane, true);
  __syncthreads();
  stage128_issue(p.WTfsc, Bbuf, tid);                  // overlap: i2 head dots
  {
    const float* h = F + lane*128;
    for (int c=0;c<3;++c) ca[c] = p.i2_b2[c];
    for (int k=0;k<128;++k){ float hv=h[k];
      for (int c=0;c<3;++c) ca[c] += hv*p.i2_w2[k*3+c]; }
  }
  wait_async(); __syncthreads();
  gemm_dump_f32(A, Bbuf, p.fs_b1, F, lane, true);
  __syncthreads();
  {
    const float* h = F + lane*128;
    lamv = p.fs_b2[0];
    for (int k=0;k<128;++k) lamv += h[k]*p.fs_w2[k];
  }

  // epilogue: per-edge vectors, masks, group-sum atomics
  if (valid){
    const int s = p.edge_index[el], r = p.edge_index[p.E + el];
    float f3[3], a3[3];
    #pragma unroll
    for (int d=0; d<3; ++d){
      f3[d] = cf[0]*va3[d] + cf[1]*vb3[d] + cf[2]*vc3[d];
      a3[d] = ca[0]*va3[d] + ca[1]*vb3[d] + ca[2]*vc3[d];
      p.fij[el*3+d] = f3[d];
      p.aij[el*3+d] = a3[d];
    }
    p.lam[el] = lamv;
    const bool isv = (p.edge_attr[el*3+0] == -1.0f);
    const bool gr  = (p.node_type[r*2+1] == -1);
    const bool gs  = (p.node_type[s*2+1] == -1);
    const bool bg  = isv && gr && !gs;
    p.wbg[el] = bg ? 1.f : 0.f;
    if (bg){
      atomicAdd(&p.cnt[r], 1.f);
      const float mwt = fmaxf(p.node_weights[s], 1e-12f);
      atomicAdd(&p.M_g[r], mwt);
      #pragma unroll
      for (int d=0; d<3; ++d){
        atomicAdd(&p.s_f[r*3+d], f3[d]);
        atomicAdd(&p.s_a[r*3+d], a3[d]);
        atomicAdd(&p.dv_sum[r*3+d], p.dv_raw[s*3+d]*mwt);
      }
    }
  }
}

// ---------------- edge pass 2 ----------------
__global__ void finalize_edges(P p)
{
  const int e = blockIdx.x*blockDim.x + threadIdx.x;
  if (e >= p.E) return;
  const int s = p.edge_index[e], r = p.edge_index[p.E + e];
  const float w  = p.wbg[e];
  const float dn = 1.f / fmaxf(p.cnt[r], 1.f);
  const float lamv = p.lam[e];
  float f3[3], a3[3], lev[3], fl[3];
  #pragma unroll
  for (int d=0; d<3; ++d){
    f3[d] = p.fij[e*3+d] - p.s_f[r*3+d]*dn*w;
    a3[d] = p.aij[e*3+d] - p.s_a[r*3+d]*dn*w;
    lev[d] = p.senders_pos[e*3+d] - p.receivers_pos[e*3+d];
    fl[d] = f3[d]*lamv;
  }
  const float mom[3] = { lev[1]*fl[2]-lev[2]*fl[1],
                         lev[2]*fl[0]-lev[0]*fl[2],
                         lev[0]*fl[1]-lev[1]*fl[0] };
  #pragma unroll
  for (int d=0; d<3; ++d){
    atomicAdd(&p.netF[r*3+d], f3[d]);
    atomicAdd(&p.netT[r*3+d], a3[d]-mom[d]);
  }
  if (w != 0.f){
    const float Mg = fmaxf(p.M_g[r], 1e-12f);
    #pragma unroll
    for (int d=0; d<3; ++d)
      p.dv_ext[s*3+d] = p.dv_raw[s*3+d] + p.dv_raw[r*3+d] - p.dv_sum[r*3+d]/Mg;
  }
}

__global__ void finalize_nodes(P p)
{
  const int i = blockIdx.x*blockDim.x + threadIdx.x;
  if (i >= p.N) return;
  const float im = p.im_dt[i], ii = p.ii_dt[i];
  #pragma unroll
  for (int d=0; d<3; ++d){
    p.out_dv[i*3+d] = im*p.netF[i*3+d] + p.dv_ext[i*3+d];
    p.out_dw[i*3+d] = ii*p.netT[i*3+d];
  }
}

__global__ void zero_kernel(float* ptr, int n){
  const int i = blockIdx.x*blockDim.x + threadIdx.x;
  if (i < n) ptr[i] = 0.f;
}

// W[K][128] row-major fp32  ->  WT[128][K] bf16
__global__ void transpose_bf16(const float* __restrict__ W, bf16_t* __restrict__ WT, int K)
{
  const int idx = blockIdx.x*blockDim.x + threadIdx.x;
  if (idx >= K*128) return;
  const int k = idx >> 7, n = idx & 127;
  WT[(size_t)n*K + k] = (bf16_t)W[idx];
}

extern "C" void kernel_launch(void* const* d_in, const int* in_sizes, int n_in,
                              void* d_out, int out_size, void* d_ws, size_t ws_size,
                              hipStream_t stream)
{
  (void)n_in; (void)out_size; (void)ws_size;
  const int N = in_sizes[12] / 128;   // node_latent [N,128]
  const int E = in_sizes[1]  / 3;     // senders_pos [E,3]

  bf16_t* WT  = (bf16_t*)d_ws;
  float*  fws = (float*)((char*)d_ws + 393216);

  P p{};
  p.edge_index    = (const int*)  d_in[0];
  p.senders_pos   = (const float*)d_in[1];
  p.receivers_pos = (const float*)d_in[2];
  p.edge_dx       = (const float*)d_in[3];
  p.edge_attr     = (const float*)d_in[4];
  p.va = (const float*)d_in[5];  p.vb = (const float*)d_in[6];  p.vc = (const float*)d_in[7];
  p.svt = (const float*)d_in[8]; p.swt = (const float*)d_in[9];
  p.rvt = (const float*)d_in[10]; p.rwt = (const float*)d_in[11];
  p.node_latent  = (const float*)d_in[12];
  p.node_type    = (const int*)  d_in[13];
  p.node_weights = (const float*)d_in[14];
  p.ef_w1=(const float*)d_in[16]; p.ef_b1=(const float*)d_in[17]; p.ef_b2=(const float*)d_in[19];
  p.ef_g =(const float*)d_in[20]; p.ef_be=(const float*)d_in[21];
  p.nf_w1=(const float*)d_in[22]; p.nf_b1=(const float*)d_in[23]; p.nf_b2=(const float*)d_in[25];
  p.nf_g =(const float*)d_in[26]; p.nf_be=(const float*)d_in[27];
  p.it_b1=(const float*)d_in[29]; p.it_b2=(const float*)d_in[31];
  p.it_g =(const float*)d_in[32]; p.it_be=(const float*)d_in[33];
  p.i1_b1=(const float*)d_in[35]; p.i1_w2=(const float*)d_in[36]; p.i1_b2=(const float*)d_in[37];
  p.i2_b1=(const float*)d_in[39]; p.i2_w2=(const float*)d_in[40]; p.i2_b2=(const float*)d_in[41];
  p.fs_b1=(const float*)d_in[43]; p.fs_w2=(const float*)d_in[44]; p.fs_b2=(const float*)d_in[45];
  p.ex_b1=(const float*)d_in[47]; p.ex_w2=(const float*)d_in[48]; p.ex_b2=(const float*)d_in[49];
  p.im_b1=(const float*)d_in[51]; p.im_w2=(const float*)d_in[52]; p.im_b2=(const float*)d_in[53];
  p.ii_b1=(const float*)d_in[55]; p.ii_w2=(const float*)d_in[56]; p.ii_b2=(const float*)d_in[57];
  p.WTnf2 = WT + 0;      p.WTef2 = WT + 16384;  p.WTint1 = WT + 32768;  p.WTint2 = WT + 81920;
  p.WTi1  = WT + 98304;  p.WTi2  = WT + 114688; p.WTfsc  = WT + 131072;
  p.WText = WT + 147456; p.WTim  = WT + 163840; p.WTii   = WT + 180224;
  float* q = fws;
  p.cnt=q;    q+=N;   p.s_f=q;    q+=3*N; p.s_a=q;    q+=3*N;
  p.M_g=q;    q+=N;   p.dv_sum=q; q+=3*N;
  p.netF=q;   q+=3*N; p.netT=q;   q+=3*N;
  p.dv_raw=q; q+=3*N; p.dv_ext=q; q+=3*N;
  p.im_dt=q;  q+=N;   p.ii_dt=q;  q+=N;
  p.fij=q;    q+=3*E; p.aij=q;    q+=3*E; p.lam=q; q+=E; p.wbg=q; q+=E;
  p.out_dv    = (float*)d_out;
  p.out_dw    = p.out_dv + (size_t)3*N;
  p.out_inter = p.out_dw + (size_t)3*N;
  p.N = N; p.E = E;

  const int zn = 17*N;
  zero_kernel<<<(zn+255)/256, 256, 0, stream>>>(p.cnt, zn);

  const struct { int src; int off; int K; } tl[10] = {
    {24,      0, 128}, {18,  16384, 128}, {28,  32768, 384}, {30,  81920, 128},
    {34,  98304, 128}, {38, 114688, 128}, {42, 131072, 128},
    {46, 147456, 128}, {50, 163840, 128}, {54, 180224, 128} };
  for (int t=0; t<10; ++t){
    const int n = tl[t].K * 128;
    transpose_bf16<<<(n+255)/256, 256, 0, stream>>>(
        (const float*)d_in[tl[t].src], WT + tl[t].off, tl[t].K);
  }

  // node MLPs: 4 waves/WG, 128 nodes per WG, 96KB LDS
  const int nwgs = (N + 127)/128;
  node_kernel<<<nwgs, 128, 98304, stream>>>(p);

  // edge pipeline: 4 waves/WG, 128 edges per WG, 160KB LDS
  const int ewgs = (E + 127)/128;
  edge_kernel<<<ewgs, 128, 163840, stream>>>(p);

  finalize_edges<<<(E+255)/256, 256, 0, stream>>>(p);
  finalize_nodes<<<(N+255)/256, 256, 0, stream>>>(p);
}
